// viscoPANN_72679436583241
// MI455X (gfx1250) — matrix operand order repaired and checked by reading, not compiled, not run on hardware
//
#include <hip/hip_runtime.h>

// ---------------------------------------------------------------------------
// viscoPANN on MI455X (gfx1250): three 128-wide MLPs per batch element.
// Dominant work = five [B x 128] * [128 x 128] GEMMs -> v_wmma_f32_16x16x32_f16
// with f32 accumulate. Layer-0 elementwise phase is vectorized x4 with packed
// f16 stores (v_cvt_pk_f16_f32 + ds_store_b64); sigmoids use v_tanh_f32 where
// available (1 trans op instead of exp+rcp).
// ---------------------------------------------------------------------------

typedef __attribute__((ext_vector_type(16))) _Float16 v16h;
typedef __attribute__((ext_vector_type(8)))  _Float16 v8h;
typedef __attribute__((ext_vector_type(4)))  _Float16 v4h;
typedef __attribute__((ext_vector_type(8)))  float    v8f;

#define WAVES  4      // waves per block
#define TILE_M 16     // batch rows per wave
#define NHID   128    // hidden width

__device__ __forceinline__ v16h make_frag(v8h a, v8h b) {
  return __builtin_shufflevector(a, b, 0,1,2,3,4,5,6,7,8,9,10,11,12,13,14,15);
}

__device__ __forceinline__ v4h pack4(float a, float b, float c, float d) {
  auto lo = __builtin_amdgcn_cvt_pkrtz(a, b);   // v_cvt_pk_f16_f32, __fp16 x2
  auto hi = __builtin_amdgcn_cvt_pkrtz(c, d);
  auto v  = __builtin_shufflevector(lo, hi, 0, 1, 2, 3);  // __fp16 x4
  return __builtin_bit_cast(v4h, v);
}

// A-fragment (16x32 f16, M=rows, K=hidden slice) from a [16][128] LDS tile.
// Contiguous K order (same as load_fragB): lane L (m=L&15, h=L>>4): elems 0..15 -> K = kt*32 + 16h + 0..15.
// Two ds_load_b128 per fragment.
__device__ __forceinline__ v16h load_fragA(const _Float16* tile, int lane, int kt) {
  const int m = lane & 15, h = lane >> 4;
  const _Float16* p = tile + m * NHID + kt * 32 + h * 16;
  v8h lo = *(const v8h*)(p);
  v8h hi = *(const v8h*)(p + 8);
  return make_frag(lo, hi);
}

// B-fragment (32x16 f16, K x N) from transposed f16 weights wT[n][k] (128x128).
// Lane L: n = nt*16 + (L&15); elems 0..15 -> K = kt*32 + 16h + 0..15.
// 32 contiguous bytes -> two global_load_b128, weight matrices stay in WGP$/L2.
__device__ __forceinline__ v16h load_fragB(const _Float16* wT, int lane, int nt, int kt) {
  const int n = nt * 16 + (lane & 15), h = lane >> 4;
  const _Float16* p = wT + n * NHID + kt * 32 + h * 16;
  v8h lo = *(const v8h*)(p);
  v8h hi = *(const v8h*)(p + 8);
  return make_frag(lo, hi);
}

__device__ __forceinline__ v8f wmma16(v16h a, v16h b, v8f c) {
  return __builtin_amdgcn_wmma_f32_16x16x32_f16(false, a, false, b, (short)0, c,
                                                false, false);
}

__device__ __forceinline__ float rcp_fast(float x) {
  return __builtin_amdgcn_rcpf(x);      // v_rcp_f32, ~1 ulp
}

__device__ __forceinline__ float softplus_f(float z) {
  return fmaxf(z, 0.f) + __logf(1.f + __expf(-fabsf(z)));
}

__device__ __forceinline__ float sigmoid_fast(float z) {
#if defined(__gfx1250__) && __has_builtin(__builtin_amdgcn_tanhf)
  // sigmoid(z) = 0.5*tanh(z/2) + 0.5 : single v_tanh_f32 trans op
  return fmaf(0.5f, __builtin_amdgcn_tanhf(0.5f * z), 0.5f);
#else
  return rcp_fast(1.f + __expf(-z));
#endif
}

// ---------------------------------------------------------------------------
// d/de of scalar MLP: e -> softplus(e*w0+b0) -> softplus(.@w1+b1) -> .@w2+b2
// Processes one 16-row tile per wave. Returns 8 per-lane row derivatives
// (rows M = r + 8*(lane>>4); identical across the 16 lanes of each half).
// ---------------------------------------------------------------------------
__device__ void mlp_grad(const float* __restrict__ e16,
                         const float* __restrict__ w0, const float* __restrict__ b0,
                         const float* __restrict__ b1, const float* __restrict__ w2,
                         const _Float16* __restrict__ wT,
                         _Float16* hT, _Float16* dT,
                         int lane, float dout[8])
{
  // layer 0 (x4 vectorized): z0 = e*w0+b0; h0 = softplus(z0); dh0 = sigmoid(z0)*w0
  for (int i = lane * 4; i < TILE_M * NHID; i += 128) {
    const int r = i >> 7, j = i & 127;
    const float  er  = e16[r];
    const float4 w0q = *(const float4*)(w0 + j);
    const float4 b0q = *(const float4*)(b0 + j);
    const float w0a[4] = {w0q.x, w0q.y, w0q.z, w0q.w};
    const float b0a[4] = {b0q.x, b0q.y, b0q.z, b0q.w};
    float sp[4], sg[4];
#pragma unroll
    for (int u = 0; u < 4; ++u) {
      const float z  = er * w0a[u] + b0a[u];
      const float ez = __expf(-fabsf(z));
      const float q  = rcp_fast(1.f + ez);         // shared reciprocal
      sp[u] = fmaxf(z, 0.f) + __logf(1.f + ez);
      sg[u] = ((z >= 0.f) ? q : ez * q) * w0a[u];
    }
    *(v4h*)(hT + i) = pack4(sp[0], sp[1], sp[2], sp[3]);   // ds_store_b64
    *(v4h*)(dT + i) = pack4(sg[0], sg[1], sg[2], sg[3]);
  }
  __syncthreads();

  v16h Ah[4], Ad[4];
#pragma unroll
  for (int kt = 0; kt < 4; ++kt) {
    Ah[kt] = load_fragA(hT, lane, kt);
    Ad[kt] = load_fragA(dT, lane, kt);
  }
#pragma unroll
  for (int r = 0; r < 8; ++r) dout[r] = 0.f;

  // layer 1 + output folded per N-tile: z1 = h0@w1, dz1 = dh0@w1 (shared B),
  // then dout += sigmoid(z1+b1) * dz1 * w2[n].
  for (int nt = 0; nt < 8; ++nt) {
    v8f accZ = {}, accD = {};
#pragma unroll
    for (int kt = 0; kt < 4; ++kt) {
      const v16h Bf = load_fragB(wT, lane, nt, kt);
      accZ = wmma16(Ah[kt], Bf, accZ);
      accD = wmma16(Ad[kt], Bf, accD);
    }
    const int   n   = nt * 16 + (lane & 15);
    const float b1n = b1[n], w2n = w2[n];
#pragma unroll
    for (int r = 0; r < 8; ++r) {
      const float s1 = sigmoid_fast(accZ[r] + b1n);
      dout[r] += s1 * accD[r] * w2n;
    }
  }
  // reduce over the 16 lanes of each half (N dimension)
#pragma unroll
  for (int m = 8; m >= 1; m >>= 1)
#pragma unroll
    for (int r = 0; r < 8; ++r) dout[r] += __shfl_xor(dout[r], m, 32);

  __syncthreads();   // tiles free for reuse by the next MLP
}

// ---------------------------------------------------------------------------
// Forward 2-input MLP (fp head): [eps,gamma] -> 128 -> 128 -> 1; returns the
// pre-b2 accumulation (h1 @ w2); caller adds b2 and applies softplus.
// ---------------------------------------------------------------------------
__device__ void mlp_fwd2(const float* __restrict__ e16, const float* __restrict__ g16,
                         const float* __restrict__ w0, const float* __restrict__ b0,
                         const float* __restrict__ b1, const float* __restrict__ w2,
                         const _Float16* __restrict__ wT, _Float16* hT,
                         int lane, float dout[8])
{
  for (int i = lane * 4; i < TILE_M * NHID; i += 128) {
    const int r = i >> 7, j = i & 127;
    const float  er  = e16[r], gr = g16[r];
    const float4 w0q = *(const float4*)(w0 + j);
    const float4 w1q = *(const float4*)(w0 + NHID + j);
    const float4 b0q = *(const float4*)(b0 + j);
    const float w0a[4] = {w0q.x, w0q.y, w0q.z, w0q.w};
    const float w1a[4] = {w1q.x, w1q.y, w1q.z, w1q.w};
    const float b0a[4] = {b0q.x, b0q.y, b0q.z, b0q.w};
    float sp[4];
#pragma unroll
    for (int u = 0; u < 4; ++u)
      sp[u] = softplus_f(er * w0a[u] + gr * w1a[u] + b0a[u]);
    *(v4h*)(hT + i) = pack4(sp[0], sp[1], sp[2], sp[3]);
  }
  __syncthreads();

  v16h Ah[4];
#pragma unroll
  for (int kt = 0; kt < 4; ++kt) Ah[kt] = load_fragA(hT, lane, kt);
#pragma unroll
  for (int r = 0; r < 8; ++r) dout[r] = 0.f;

  for (int nt = 0; nt < 8; ++nt) {
    v8f accZ = {};
#pragma unroll
    for (int kt = 0; kt < 4; ++kt) {
      const v16h Bf = load_fragB(wT, lane, nt, kt);
      accZ = wmma16(Ah[kt], Bf, accZ);
    }
    const int   n   = nt * 16 + (lane & 15);
    const float b1n = b1[n], w2n = w2[n];
#pragma unroll
    for (int r = 0; r < 8; ++r) dout[r] += softplus_f(accZ[r] + b1n) * w2n;
  }
#pragma unroll
  for (int m = 8; m >= 1; m >>= 1)
#pragma unroll
    for (int r = 0; r < 8; ++r) dout[r] += __shfl_xor(dout[r], m, 32);
}

// ---------------------------------------------------------------------------
__launch_bounds__(WAVES * 32)
__global__ void viscoPANN_main(
    const float* __restrict__ inp, const float* __restrict__ gam,
    const float* __restrict__ wv_w0, const float* __restrict__ wv_b0,
    const float* __restrict__ wv_b1, const float* __restrict__ wv_w2,
    const float* __restrict__ we_w0, const float* __restrict__ we_b0,
    const float* __restrict__ we_b1, const float* __restrict__ we_w2,
    const float* __restrict__ fp_w0, const float* __restrict__ fp_b0,
    const float* __restrict__ fp_b1, const float* __restrict__ fp_w2,
    const float* __restrict__ fp_b2,
    const _Float16* __restrict__ wvT, const _Float16* __restrict__ weT,
    const _Float16* __restrict__ fpT,
    float* __restrict__ out_sigma, float* __restrict__ out_gnew)
{
  __shared__ __align__(16) _Float16 hA[WAVES][TILE_M][NHID];
  __shared__ __align__(16) _Float16 dA[WAVES][TILE_M][NHID];
  __shared__ float sEps[WAVES][TILE_M];
  __shared__ float sDt [WAVES][TILE_M];
  __shared__ float sGam[WAVES][TILE_M];
  __shared__ float sE  [WAVES][TILE_M];

  const int lane = threadIdx.x & 31;
  const int wid  = threadIdx.x >> 5;
  const int half = lane >> 4;
  const int rowbase = (blockIdx.x * WAVES + wid) * TILE_M;

  if (lane < TILE_M) {
    const int row = rowbase + lane;
    const float e = inp[2 * row];
    sEps[wid][lane] = e;
    sDt [wid][lane] = inp[2 * row + 1];
    const float g = gam[row];
    sGam[wid][lane] = g;
    sE  [wid][lane] = e - g;       // Wv input: eps - gamma
  }
  __syncthreads();

  float dWv[8], dWe[8], fpacc[8];

  mlp_grad(sE[wid],   wv_w0, wv_b0, wv_b1, wv_w2, wvT,
           &hA[wid][0][0], &dA[wid][0][0], lane, dWv);
  mlp_grad(sEps[wid], we_w0, we_b0, we_b1, we_w2, weT,
           &hA[wid][0][0], &dA[wid][0][0], lane, dWe);
  mlp_fwd2(sEps[wid], sGam[wid], fp_w0, fp_b0, fp_b1, fp_w2, fpT,
           &hA[wid][0][0], lane, fpacc);

  const float fpb2 = fp_b2[0];
  if ((lane & 15) == 0) {          // lanes 0 (rows 0-7) and 16 (rows 8-15)
#pragma unroll
    for (int r = 0; r < 8; ++r) {
      const int   m   = r + 8 * half;
      const int   row = rowbase + m;
      const float fpv = softplus_f(fpacc[r] + fpb2);
      const float dwv = dWv[r];
      out_sigma[row] = dwv + dWe[r];
      out_gnew [row] = sGam[wid][m] + sDt[wid][m] * fpv * dwv;
    }
  }
}

// Pre-transpose + f32->f16 convert the three 128x128 hidden weights into d_ws:
// dst[mat][n][k] = w1_mat[k][n], so B-fragments are contiguous along K.
__global__ void prep_weights(const float* __restrict__ w1a,
                             const float* __restrict__ w1b,
                             const float* __restrict__ w1c,
                             _Float16* __restrict__ dst)
{
  const int idx = blockIdx.x * blockDim.x + threadIdx.x;   // 0 .. 3*16384-1
  const int mat = idx >> 14;
  const int loc = idx & 16383;
  const int n = loc >> 7, k = loc & 127;
  const float* src = (mat == 0) ? w1a : (mat == 1) ? w1b : w1c;
  dst[idx] = (_Float16)src[k * NHID + n];
}

extern "C" void kernel_launch(void* const* d_in, const int* in_sizes, int n_in,
                              void* d_out, int out_size, void* d_ws, size_t ws_size,
                              hipStream_t stream)
{
  const float* inputs = (const float*)d_in[0];
  const float* gamma  = (const float*)d_in[1];
  const float* wv_w0  = (const float*)d_in[2];
  const float* wv_w1  = (const float*)d_in[3];
  const float* wv_w2  = (const float*)d_in[4];
  const float* wv_b0  = (const float*)d_in[5];
  const float* wv_b1  = (const float*)d_in[6];
  const float* we_w0  = (const float*)d_in[8];
  const float* we_w1  = (const float*)d_in[9];
  const float* we_w2  = (const float*)d_in[10];
  const float* we_b0  = (const float*)d_in[11];
  const float* we_b1  = (const float*)d_in[12];
  const float* fp_w0  = (const float*)d_in[14];
  const float* fp_w1  = (const float*)d_in[15];
  const float* fp_w2  = (const float*)d_in[16];
  const float* fp_b0  = (const float*)d_in[17];
  const float* fp_b1  = (const float*)d_in[18];
  const float* fp_b2  = (const float*)d_in[19];

  const int B = in_sizes[0] / 2;          // 131072
  _Float16* wT = (_Float16*)d_ws;         // 3 * 128*128 f16 = 96 KB scratch

  prep_weights<<<(3 * NHID * NHID) / 256, 256, 0, stream>>>(wv_w1, we_w1, fp_w1, wT);

  const int blocks = B / (TILE_M * WAVES);  // 131072 / 64 = 2048
  viscoPANN_main<<<blocks, WAVES * 32, 0, stream>>>(
      inputs, gamma,
      wv_w0, wv_b0, wv_b1, wv_w2,
      we_w0, we_b0, we_b1, we_w2,
      fp_w0, fp_b0, fp_b1, fp_w2, fp_b2,
      wT, wT + NHID * NHID, wT + 2 * NHID * NHID,
      (float*)d_out, (float*)d_out + B);
}
